// CameraHead_16509854285951
// MI455X (gfx1250) — matrix operand order repaired and checked by reading, not compiled, not run on hardware
//
#include <hip/hip_runtime.h>

#define TOK 65536   // 16*16*256 tokens
#define DD  512

typedef __attribute__((ext_vector_type(16))) __bf16       v16bf;
typedef __attribute__((ext_vector_type(8)))  float        v8f;
typedef __attribute__((ext_vector_type(4)))  unsigned int u32x4;

union Frag { v16bf v; u32x4 q[2]; };

// ---------------------------------------------------------------------------
// f32 -> bf16 conversion
// ---------------------------------------------------------------------------
__global__ void f32_to_bf16_kernel(const float* __restrict__ src,
                                   __bf16* __restrict__ dst, int n) {
    int i = blockIdx.x * blockDim.x + threadIdx.x;
    if (i < n) dst[i] = (__bf16)src[i];
}

// ---------------------------------------------------------------------------
// Fused GEMM: out[m,n] = relu( sum_k A[m,k]*W[n,k] + bias[n] )  (+ residual)
// M = TOK (grid.y*128), N = K = 512.  bf16 inputs, f32 WMMA accumulate.
// WG: 256 threads = 8 waves as 4(M) x 2(N); wave tile 32x32 (2x2 WMMA tiles).
// ---------------------------------------------------------------------------
template <bool RES>
__global__ __launch_bounds__(256)
void gemm512_relu(const __bf16* __restrict__ A, const __bf16* __restrict__ W,
                  const float* __restrict__ bias,
                  const __bf16* __restrict__ xres, __bf16* __restrict__ out) {
    const int tid    = threadIdx.x;
    const int lane   = tid & 31;
    const int laneLo = lane & 15;
    const int laneHi = lane >> 4;
    const int wave   = tid >> 5;                       // 0..7
    const int m_wave = blockIdx.y * 128 + (wave & 3) * 32;
    const int n_wave = blockIdx.x * 64  + (wave >> 2) * 32;

    // accumulators pre-loaded with bias (C/D layout: n = laneLo, m = v + 8*laneHi)
    const float b0 = bias[n_wave + laneLo];
    const float b1 = bias[n_wave + 16 + laneLo];
    v8f c[2][2];
#pragma unroll
    for (int i = 0; i < 8; ++i) {
        c[0][0][i] = b0; c[1][0][i] = b0;
        c[0][1][i] = b1; c[1][1][i] = b1;
    }

    // A-frag: lane<16 -> K 0..15 contiguous; lane>=16 -> K 16..31
    const __bf16* a0p = A + (size_t)(m_wave + laneLo) * DD + laneHi * 16;
    const __bf16* a1p = a0p + (size_t)16 * DD;
    // B-frag: lane<16 -> K 0..15 contiguous; lane>=16 -> K 16..31 (col = laneLo)
    const __bf16* b0p = W + (size_t)(n_wave + laneLo) * DD + laneHi * 16;
    const __bf16* b1p = b0p + (size_t)16 * DD;

    for (int k0 = 0; k0 < DD; k0 += 32) {
        Frag a0, a1, bb0, bb1;
        a0.q[0] = *(const u32x4*)(a0p + k0);
        a0.q[1] = *(const u32x4*)(a0p + k0 + 8);
        a1.q[0] = *(const u32x4*)(a1p + k0);
        a1.q[1] = *(const u32x4*)(a1p + k0 + 8);
        bb0.v   = *(const v16bf*)(b0p + k0);
        bb1.v   = *(const v16bf*)(b1p + k0);
        c[0][0] = __builtin_amdgcn_wmma_f32_16x16x32_bf16(false, a0.v, false, bb0.v,
                                                          (short)0, c[0][0], false, false);
        c[0][1] = __builtin_amdgcn_wmma_f32_16x16x32_bf16(false, a0.v, false, bb1.v,
                                                          (short)0, c[0][1], false, false);
        c[1][0] = __builtin_amdgcn_wmma_f32_16x16x32_bf16(false, a1.v, false, bb0.v,
                                                          (short)0, c[1][0], false, false);
        c[1][1] = __builtin_amdgcn_wmma_f32_16x16x32_bf16(false, a1.v, false, bb1.v,
                                                          (short)0, c[1][1], false, false);
    }

#pragma unroll
    for (int mi = 0; mi < 2; ++mi)
#pragma unroll
        for (int ni = 0; ni < 2; ++ni)
#pragma unroll
            for (int v = 0; v < 8; ++v) {
                const int m = m_wave + mi * 16 + laneHi * 8 + v;
                const int n = n_wave + ni * 16 + laneLo;
                float r = fmaxf((float)c[mi][ni][v], 0.0f);
                if (RES) r += (float)xres[(size_t)m * DD + n];
                out[(size_t)m * DD + n] = (__bf16)r;
            }
}

// ---------------------------------------------------------------------------
// Mean over n=256 tokens per (b*v) group -> pooled (256 x 512) f32
// ---------------------------------------------------------------------------
__global__ void pool_kernel(const __bf16* __restrict__ xb, float* __restrict__ pooled) {
    const int g = blockIdx.x;       // 0..255
    const int d = threadIdx.x;      // 0..255 -> handles d and d+256
    const __bf16* base = xb + (size_t)g * 256 * DD;
    float s0 = 0.f, s1 = 0.f;
    for (int i = 0; i < 256; ++i) {
        s0 += (float)base[(size_t)i * DD + d];
        s1 += (float)base[(size_t)i * DD + d + 256];
    }
    pooled[g * DD + d]       = s0 * (1.0f / 256.0f);
    pooled[g * DD + d + 256] = s1 * (1.0f / 256.0f);
}

// ---------------------------------------------------------------------------
// Small dense layer (M=256, N=K=512) in f32: y = relu(x W^T + b)
// ---------------------------------------------------------------------------
__global__ void dense512_relu(const float* __restrict__ x, const float* __restrict__ w,
                              const float* __restrict__ b, float* __restrict__ y) {
    __shared__ float xs[DD];
    const int m = blockIdx.x;
    for (int i = threadIdx.x; i < DD; i += blockDim.x) xs[i] = x[(size_t)m * DD + i];
    __syncthreads();
    for (int n = threadIdx.x; n < DD; n += blockDim.x) {
        const float* wr = w + (size_t)n * DD;
        float acc = b[n];
        for (int k = 0; k < DD; ++k) acc = fmaf(xs[k], wr[k], acc);
        y[(size_t)m * DD + n] = fmaxf(acc, 0.0f);
    }
}

// ---------------------------------------------------------------------------
// Heads: t (3) and rot (9) per row
// ---------------------------------------------------------------------------
__global__ void heads_kernel(const float* __restrict__ f,
                             const float* __restrict__ tw, const float* __restrict__ tb,
                             const float* __restrict__ rw, const float* __restrict__ rb,
                             float* __restrict__ tvec, float* __restrict__ rotm) {
    const int m = blockIdx.x;
    const int j = threadIdx.x;
    if (j >= 12) return;
    const float* wr;
    float acc;
    if (j < 3) { wr = tw + (size_t)j * DD;       acc = tb[j];     }
    else       { wr = rw + (size_t)(j - 3) * DD; acc = rb[j - 3]; }
    const float* xr = f + (size_t)m * DD;
    for (int k = 0; k < DD; ++k) acc = fmaf(xr[k], wr[k], acc);
    if (j < 3) tvec[m * 3 + j] = acc;
    else       rotm[m * 9 + (j - 3)] = acc;
}

// ---------------------------------------------------------------------------
// Row-normalize 3x3, project onto SO(3) via Horn quaternion (== SVD + det fix),
// assemble 4x4 pose.
// ---------------------------------------------------------------------------
__global__ void pose_kernel(const float* __restrict__ tvec, const float* __restrict__ rotm,
                            float* __restrict__ out) {
    const int m = blockIdx.x * blockDim.x + threadIdx.x;
    if (m >= 256) return;

    float Mx[3][3];
    for (int r = 0; r < 3; ++r) {
        float a = rotm[m * 9 + r * 3 + 0];
        float b = rotm[m * 9 + r * 3 + 1];
        float cc = rotm[m * 9 + r * 3 + 2];
        float nrm = fmaxf(sqrtf(a * a + b * b + cc * cc), 1e-12f);
        float inv = 1.0f / nrm;
        Mx[r][0] = a * inv; Mx[r][1] = b * inv; Mx[r][2] = cc * inv;
    }
    // Kearsley/Horn 4x4 for R = argmax tr(R^T M) with det(R)=+1
    const float K00 = Mx[0][0] + Mx[1][1] + Mx[2][2];
    const float K01 = Mx[2][1] - Mx[1][2];
    const float K02 = Mx[0][2] - Mx[2][0];
    const float K03 = Mx[1][0] - Mx[0][1];
    const float K11 = Mx[0][0] - Mx[1][1] - Mx[2][2];
    const float K12 = Mx[0][1] + Mx[1][0];
    const float K13 = Mx[2][0] + Mx[0][2];
    const float K22 = -Mx[0][0] + Mx[1][1] - Mx[2][2];
    const float K23 = Mx[1][2] + Mx[2][1];
    const float K33 = -Mx[0][0] - Mx[1][1] + Mx[2][2];

    const float sh = 4.0f;   // rows normalized -> |eig| <= 3; shift makes top eig dominant
    float q0 = 1.0f, q1 = 0.1f, q2 = 0.2f, q3 = 0.3f;
    for (int it = 0; it < 100; ++it) {
        float p0 = (K00 + sh) * q0 + K01 * q1 + K02 * q2 + K03 * q3;
        float p1 = K01 * q0 + (K11 + sh) * q1 + K12 * q2 + K13 * q3;
        float p2 = K02 * q0 + K12 * q1 + (K22 + sh) * q2 + K23 * q3;
        float p3 = K03 * q0 + K13 * q1 + K23 * q2 + (K33 + sh) * q3;
        float inv = rsqrtf(p0 * p0 + p1 * p1 + p2 * p2 + p3 * p3 + 1e-30f);
        q0 = p0 * inv; q1 = p1 * inv; q2 = p2 * inv; q3 = p3 * inv;
    }
    const float w = q0, x = q1, y = q2, z = q3;
    float R[3][3];
    R[0][0] = 1 - 2 * (y * y + z * z); R[0][1] = 2 * (x * y - w * z); R[0][2] = 2 * (x * z + w * y);
    R[1][0] = 2 * (x * y + w * z); R[1][1] = 1 - 2 * (x * x + z * z); R[1][2] = 2 * (y * z - w * x);
    R[2][0] = 2 * (x * z - w * y); R[2][1] = 2 * (y * z + w * x); R[2][2] = 1 - 2 * (x * x + y * y);

    float* o = out + (size_t)m * 16;
    for (int r = 0; r < 3; ++r) {
        o[r * 4 + 0] = R[r][0];
        o[r * 4 + 1] = R[r][1];
        o[r * 4 + 2] = R[r][2];
        o[r * 4 + 3] = tvec[m * 3 + r];
    }
    o[12] = 0.0f; o[13] = 0.0f; o[14] = 0.0f; o[15] = 1.0f;
}

// ---------------------------------------------------------------------------
extern "C" void kernel_launch(void* const* d_in, const int* in_sizes, int n_in,
                              void* d_out, int out_size, void* d_ws, size_t ws_size,
                              hipStream_t stream) {
    const float* feat = (const float*)d_in[0];
    const float* rw[6] = { (const float*)d_in[1], (const float*)d_in[3], (const float*)d_in[5],
                           (const float*)d_in[7], (const float*)d_in[9], (const float*)d_in[11] };
    const float* rb[6] = { (const float*)d_in[2], (const float*)d_in[4], (const float*)d_in[6],
                           (const float*)d_in[8], (const float*)d_in[10], (const float*)d_in[12] };
    const float* m_w1 = (const float*)d_in[13];
    const float* m_b1 = (const float*)d_in[14];
    const float* m_w2 = (const float*)d_in[15];
    const float* m_b2 = (const float*)d_in[16];
    const float* t_w  = (const float*)d_in[17];
    const float* t_b  = (const float*)d_in[18];
    const float* rot_w = (const float*)d_in[19];
    const float* rot_b = (const float*)d_in[20];

    // ---- workspace carve ----
    const size_t NTOK = (size_t)TOK * DD;        // 33,554,432 elements
    char* ws = (char*)d_ws;
    __bf16* XB = (__bf16*)ws;                    // 64 MB
    __bf16* H1 = XB + NTOK;                      // 64 MB
    __bf16* H2 = H1 + NTOK;                      // 64 MB
    __bf16* WB = H2 + NTOK;                      // 6 * 512*512 bf16 = 3 MB
    float* pooled = (float*)(WB + 6 * (size_t)DD * DD);
    float* f1   = pooled + 256 * DD;
    float* f2   = f1 + 256 * DD;
    float* tv   = f2 + 256 * DD;
    float* rotm = tv + 256 * 3;

    // ---- convert inputs to bf16 ----
    f32_to_bf16_kernel<<<(int)((NTOK + 255) / 256), 256, 0, stream>>>(feat, XB, (int)NTOK);
    for (int i = 0; i < 6; ++i)
        f32_to_bf16_kernel<<<(DD * DD + 255) / 256, 256, 0, stream>>>(
            rw[i], WB + (size_t)i * DD * DD, DD * DD);

    // ---- 6 WMMA GEMM layers (two residual blocks) ----
    const dim3 ggrid(DD / 64, TOK / 128);   // (8, 512)
    gemm512_relu<false><<<ggrid, 256, 0, stream>>>(XB, WB + 0 * (size_t)DD * DD, rb[0], nullptr, H1);
    gemm512_relu<false><<<ggrid, 256, 0, stream>>>(H1, WB + 1 * (size_t)DD * DD, rb[1], nullptr, H2);
    gemm512_relu<true ><<<ggrid, 256, 0, stream>>>(H2, WB + 2 * (size_t)DD * DD, rb[2], XB, XB);
    gemm512_relu<false><<<ggrid, 256, 0, stream>>>(XB, WB + 3 * (size_t)DD * DD, rb[3], nullptr, H1);
    gemm512_relu<false><<<ggrid, 256, 0, stream>>>(H1, WB + 4 * (size_t)DD * DD, rb[4], nullptr, H2);
    gemm512_relu<true ><<<ggrid, 256, 0, stream>>>(H2, WB + 5 * (size_t)DD * DD, rb[5], XB, XB);

    // ---- pool + small MLPs + heads + pose ----
    pool_kernel<<<256, 256, 0, stream>>>(XB, pooled);
    dense512_relu<<<256, 128, 0, stream>>>(pooled, m_w1, m_b1, f1);
    dense512_relu<<<256, 128, 0, stream>>>(f1, m_w2, m_b2, f2);
    heads_kernel<<<256, 32, 0, stream>>>(f2, t_w, t_b, rot_w, rot_b, tv, rotm);
    pose_kernel<<<8, 32, 0, stream>>>(tv, rotm, (float*)d_out);
}
